// EmbeddingLSTM_65627100282898
// MI455X (gfx1250) — matrix operand order repaired and checked by reading, not compiled, not run on hardware
//
#include <hip/hip_runtime.h>

typedef __bf16 bf16;
typedef __attribute__((ext_vector_type(16))) __bf16 v16bf;
typedef __attribute__((ext_vector_type(8)))  float  v8f;

#define B_   64
#define F_   256
#define T_   512
#define SP_  512          // 2F
#define H_   1024
#define KTOT 1536         // SP + H
#define NTOT 4096         // 4 gates * H, interleaved n = 4*j + g
#define KT_N (KTOT / 32)  // 48 k-tiles
#define NWG  128

#define WLDS_BYTES (2 * KT_N * 512 * 2)   // 96 KB per-WG weight slice
#define SMEM_BYTES (64 * 32 * 4)          // 8 KB pre-activation slice
#define DYN_LDS    (WLDS_BYTES + SMEM_BYTES)

// ---------------- workspace layout (bytes) ----------------
#define OFF_BAR 0u
#define OFF_H   256u                               // 2 x [64][1024] bf16 = 256 KB
#define OFF_W   (OFF_H + 2u*64u*1024u*2u)          // packed W bf16 = 12.6 MB
#define OFF_XIN (OFF_W + (unsigned)KTOT*NTOT*2u)   // [T][B][512] bf16 = 33.5 MB

// Pack W (4 gates, each [1536,1024] row-major [k][j], fp32) into bf16
// B-fragment tiles. Global column n = 4*j + g. Fragment (nt, kt) is 32x16:
// lane l holds K = kt*32 + (l/16)*8 + (s&7) + (s>>3)*16 (s=0..15), N = nt*16 + (l%16),
// stored contiguously: frag_base + l*16 + s.
__global__ void pack_w(const float* __restrict__ Wf, const float* __restrict__ Wi,
                       const float* __restrict__ Wu, const float* __restrict__ Wo,
                       bf16* __restrict__ dst) {
    int idx = blockIdx.x * 256 + threadIdx.x;     // over KTOT*NTOT
    int k = idx >> 12;
    int n = idx & 4095;
    int j = n >> 2;
    int g = n & 3;
    const float* W = (g == 0) ? Wf : (g == 1) ? Wi : (g == 2) ? Wu : Wo;
    float v = W[k * H_ + j];
    int nt   = n >> 4;
    int lane = (n & 15) | (((k >> 3) & 1) << 4);
    int slot = (k & 7) | (((k >> 4) & 1) << 3);
    int kt   = k >> 5;
    dst[((nt * KT_N + kt) << 9) + lane * 16 + slot] = (bf16)v;
}

// Interleaved input: xin[t][b][2f]   = x[b][f][t]
//                    xin[t][b][2f+1] = (t>0) ? y[b][f][t-1] : 0
__global__ void make_xin(const float* __restrict__ x, const float* __restrict__ y,
                         bf16* __restrict__ dst) {
    long idx = (long)blockIdx.x * 256 + threadIdx.x;   // over T*B*SP
    int  kk = (int)(idx & 511);
    long tb = idx >> 9;
    int  b  = (int)(tb & 63);
    int  t  = (int)(tb >> 6);
    int  f  = kk >> 1;
    float v;
    if ((kk & 1) == 0) v = x[((long)b * F_ + f) * T_ + t];
    else               v = (t > 0) ? y[((long)b * F_ + f) * T_ + (t - 1)] : 0.f;
    dst[idx] = (bf16)v;
}

__device__ __forceinline__ float sigmoidf_(float v) {
    return 1.f / (1.f + __expf(-v));
}

// Persistent recurrent kernel. 128 WGs x 8 waves. WG wg owns global n-tiles
// {2wg, 2wg+1} (hidden units j in [wg*8, wg*8+8), all 4 gates), all 4
// m-tiles. Wave w: m-tile = w>>1, local n-tile = w&1. Weights live in LDS.
__launch_bounds__(256)
__global__ void lstm_persistent(const bf16* __restrict__ xin,
                                const bf16* __restrict__ Wp,
                                const float* __restrict__ bF, const float* __restrict__ bI,
                                const float* __restrict__ bU, const float* __restrict__ bO,
                                bf16* __restrict__ hbuf,
                                unsigned* __restrict__ bar,
                                float* __restrict__ out) {
    extern __shared__ char dynsm[];
    bf16*  wlds = (bf16*)dynsm;                      // 96 KB weight slice
    float* smem = (float*)(dynsm + WLDS_BYTES);      // 8 KB pre-activations

    const int wg   = blockIdx.x;
    const int tid  = threadIdx.x;
    const int wave = tid >> 5;
    const int lane = tid & 31;
    const int mt   = wave >> 1;                 // 0..3
    const int nl   = wave & 1;                  // 0..1
    const int rowA = mt * 16 + (lane & 15);     // batch row for A frag
    const int aoff = (lane < 16) ? 0 : 8;       // K-half select for A

    // ---- stage this WG's weight slice (global -> LDS) via async-to-LDS ----
    {
        const char* gsrc = (const char*)(Wp + (((size_t)wg * 2) * KT_N << 9));
        for (int i = tid; i < WLDS_BYTES / 16; i += 256) {
            unsigned           laddr = (unsigned)(size_t)(void*)((char*)wlds + (size_t)i * 16);
            unsigned long long gaddr = (unsigned long long)(size_t)(gsrc + (size_t)i * 16);
            asm volatile("global_load_async_to_lds_b128 %0, %1, off"
                         :: "v"(laddr), "v"(gaddr) : "memory");
        }
        asm volatile("s_wait_asynccnt 0x0" ::: "memory");
    }
    __syncthreads();

    // ---- per-thread fixed (row, j) ownership for the elementwise tail ----
    const int jl = tid & 7;
    const int j  = wg * 8 + jl;
    const int r0 = tid >> 3;        // rows r0 and r0+32
    const float Bf = bF[j], Bi = bI[j], Bu = bU[j], Bo = bO[j];
    float c0r = 0.f, c1r = 0.f;     // cell state lives in registers

    unsigned* cnt = bar;
    unsigned* gen = bar + 1;

    for (int t = 0; t < T_; ++t) {
        const bf16* hb   = hbuf + (size_t)(t & 1) * (B_ * H_);
        bf16*       hbn  = hbuf + (size_t)((t + 1) & 1) * (B_ * H_);
        const bf16* xrow = xin + ((long)t * B_ + rowA) * SP_;
        const bf16* hrow = hb + (long)rowA * H_;

        if (t + 1 < T_) __builtin_prefetch(xrow + (size_t)B_ * SP_, 0, 1);

        v8f acc0 = {}, acc1 = {};
        // K in [0, 512): input projection part (2 accumulators for ILP)
        #pragma unroll 2
        for (int kt = 0; kt < SP_ / 32; kt += 2) {
            union { v16bf v; uint4 u[2]; } A0, A1, B0, B1;
            const bf16* ap0 = xrow + kt * 32 + aoff;
            const bf16* ap1 = ap0 + 32;
            A0.u[0] = *(const uint4*)(ap0);
            A0.u[1] = *(const uint4*)(ap0 + 16);
            A1.u[0] = *(const uint4*)(ap1);
            A1.u[1] = *(const uint4*)(ap1 + 16);
            const bf16* bp0 = wlds + ((nl * KT_N + kt) << 9) + lane * 16;
            const bf16* bp1 = bp0 + 512;
            B0.u[0] = *(const uint4*)(bp0);
            B0.u[1] = *(const uint4*)(bp0 + 8);
            B1.u[0] = *(const uint4*)(bp1);
            B1.u[1] = *(const uint4*)(bp1 + 8);
            acc0 = __builtin_amdgcn_wmma_f32_16x16x32_bf16(
                false, A0.v, false, B0.v, (short)0, acc0, false, false);
            acc1 = __builtin_amdgcn_wmma_f32_16x16x32_bf16(
                false, A1.v, false, B1.v, (short)0, acc1, false, false);
        }
        // K in [512, 1536): recurrent part
        #pragma unroll 2
        for (int kt = SP_ / 32; kt < KT_N; kt += 2) {
            union { v16bf v; uint4 u[2]; } A0, A1, B0, B1;
            const bf16* ap0 = hrow + (kt * 32 - SP_) + aoff;
            const bf16* ap1 = ap0 + 32;
            A0.u[0] = *(const uint4*)(ap0);
            A0.u[1] = *(const uint4*)(ap0 + 16);
            A1.u[0] = *(const uint4*)(ap1);
            A1.u[1] = *(const uint4*)(ap1 + 16);
            const bf16* bp0 = wlds + ((nl * KT_N + kt) << 9) + lane * 16;
            const bf16* bp1 = bp0 + 512;
            B0.u[0] = *(const uint4*)(bp0);
            B0.u[1] = *(const uint4*)(bp0 + 8);
            B1.u[0] = *(const uint4*)(bp1);
            B1.u[1] = *(const uint4*)(bp1 + 8);
            acc0 = __builtin_amdgcn_wmma_f32_16x16x32_bf16(
                false, A0.v, false, B0.v, (short)0, acc0, false, false);
            acc1 = __builtin_amdgcn_wmma_f32_16x16x32_bf16(
                false, A1.v, false, B1.v, (short)0, acc1, false, false);
        }
        acc0 += acc1;

        // spill C fragment to LDS: C frag VGPR v -> row mt*16 + v + (lane<16?0:8)
        {
            int colL  = nl * 16 + (lane & 15);
            int rbase = mt * 16 + ((lane < 16) ? 0 : 8);
            #pragma unroll
            for (int v = 0; v < 8; ++v)
                smem[(rbase + v) * 32 + colL] = acc0[v];
        }
        __syncthreads();

        // fused gate elementwise: rows r0 and r0+32, hidden unit j
        {
            float pf = smem[r0 * 32 + jl * 4 + 0] + Bf;
            float pi = smem[r0 * 32 + jl * 4 + 1] + Bi;
            float pu = smem[r0 * 32 + jl * 4 + 2] + Bu;
            float po = smem[r0 * 32 + jl * 4 + 3] + Bo;
            float cn = c0r * sigmoidf_(pf) + sigmoidf_(pi) * tanhf(pu);
            float hn = sigmoidf_(po) * tanhf(cn);
            c0r = cn;
            hbn[r0 * H_ + j] = (bf16)hn;
            out[((long)r0 * H_ + j) * T_ + t] = hn;
        }
        {
            int r1 = r0 + 32;
            float pf = smem[r1 * 32 + jl * 4 + 0] + Bf;
            float pi = smem[r1 * 32 + jl * 4 + 1] + Bi;
            float pu = smem[r1 * 32 + jl * 4 + 2] + Bu;
            float po = smem[r1 * 32 + jl * 4 + 3] + Bo;
            float cn = c1r * sigmoidf_(pf) + sigmoidf_(pi) * tanhf(pu);
            float hn = sigmoidf_(po) * tanhf(cn);
            c1r = cn;
            hbn[r1 * H_ + j] = (bf16)hn;
            out[((long)r1 * H_ + j) * T_ + t] = hn;
        }
        __syncthreads();
        __threadfence();

        // device-wide phase barrier
        if (tid == 0) {
            unsigned g = __atomic_load_n(gen, __ATOMIC_ACQUIRE);
            if (atomicAdd(cnt, 1u) == NWG - 1) {
                __atomic_store_n(cnt, 0u, __ATOMIC_RELAXED);
                __atomic_store_n(gen, g + 1, __ATOMIC_RELEASE);
            } else {
                while (__atomic_load_n(gen, __ATOMIC_ACQUIRE) == g)
                    __builtin_amdgcn_s_sleep(2);
            }
        }
        __syncthreads();
        __threadfence();
    }

    // final cell state c_fin -> out[B*H*T ...]
    out[(long)B_ * H_ * T_ + (long)r0 * H_ + j]        = c0r;
    out[(long)B_ * H_ * T_ + (long)(r0 + 32) * H_ + j] = c1r;
}

extern "C" void kernel_launch(void* const* d_in, const int* in_sizes, int n_in,
                              void* d_out, int out_size, void* d_ws, size_t ws_size,
                              hipStream_t stream) {
    const float* x   = (const float*)d_in[0];
    const float* y   = (const float*)d_in[1];
    // d_in[2] subject_id, d_in[11] We, d_in[12] be: unused in 'none' mode
    const float* Wf  = (const float*)d_in[3];
    const float* bf_ = (const float*)d_in[4];
    const float* Wi  = (const float*)d_in[5];
    const float* bi_ = (const float*)d_in[6];
    const float* Wu  = (const float*)d_in[7];
    const float* bu_ = (const float*)d_in[8];
    const float* Wo  = (const float*)d_in[9];
    const float* bo_ = (const float*)d_in[10];

    char*     ws   = (char*)d_ws;
    unsigned* bar  = (unsigned*)(ws + OFF_BAR);
    bf16*     hbuf = (bf16*)(ws + OFF_H);
    bf16*     Wp   = (bf16*)(ws + OFF_W);
    bf16*     xin  = (bf16*)(ws + OFF_XIN);
    float*    out  = (float*)d_out;

    // zero barrier + h0 (both phases)
    (void)hipMemsetAsync(ws, 0, OFF_W, stream);

    pack_w  <<<(KTOT * NTOT) / 256, 256, 0, stream>>>(Wf, Wi, Wu, Wo, Wp);
    make_xin<<<((long)T_ * B_ * SP_) / 256, 256, 0, stream>>>(x, y, xin);
    lstm_persistent<<<NWG, 256, DYN_LDS, stream>>>(xin, Wp, bf_, bi_, bu_, bo_,
                                                   hbuf, bar, out);
}
